// SeparatedGCNModelVAE_88502096101452
// MI455X (gfx1250) — matrix operand order repaired and checked by reading, not compiled, not run on hardware
//
#include <hip/hip_runtime.h>
#include <math.h>

typedef __attribute__((ext_vector_type(16))) __bf16 v16bf;
typedef __attribute__((ext_vector_type(8)))  float  v8f;

union FragAB { unsigned u[8]; v16bf v; };

__device__ __forceinline__ unsigned short f2bf_bits(float f) {
  union { float fv; unsigned uv; } c; c.fv = f;
  unsigned u = c.uv;
  u += 0x7fffu + ((u >> 16) & 1u);   // RNE truncation to bf16
  return (unsigned short)(u >> 16);
}

// ----------------------------- elementwise ---------------------------------
__global__ void k_zero_f32(float* __restrict__ p, size_t n) {
  size_t i = (size_t)blockIdx.x * blockDim.x + threadIdx.x;
  if (i < n) p[i] = 0.f;
}

__global__ void k_to_bf16(const float* __restrict__ in, unsigned short* __restrict__ out,
                          size_t n, int relu) {
  size_t i = (size_t)blockIdx.x * blockDim.x + threadIdx.x;
  if (i < n) {
    float v = in[i];
    if (relu) v = fmaxf(v, 0.f);
    out[i] = f2bf_bits(v);
  }
}

__global__ void k_clip_write(const float* __restrict__ in, float* __restrict__ out, size_t n) {
  size_t i = (size_t)blockIdx.x * blockDim.x + threadIdx.x;
  if (i < n) out[i] = fminf(fmaxf(in[i], -5.f), 2.f);
}

__global__ void k_write_z(const float* __restrict__ zm, float* __restrict__ oz,
                          float* __restrict__ ozm, unsigned short* __restrict__ zbf, size_t n) {
  size_t i = (size_t)blockIdx.x * blockDim.x + threadIdx.x;
  if (i < n) {
    float v = zm[i];
    oz[i]  = v;
    ozm[i] = v;
    zbf[i] = f2bf_bits(v);
  }
}

// -------- pack f32 weight W[K x Nout] into WMMA B-fragment layout -----------
// dword index = ((tn*(K/32) + ks)*32 + lane)*8 + v
// content: bf16 pair at k = ks*32 + 2v + 8*(lane>>4) + 8*(v>>2), n = tn*16 + (lane&15)
__global__ void k_pack_b(const float* __restrict__ W, unsigned* __restrict__ P,
                         int K, int Nout) {
  int total = (K >> 5) * (Nout >> 4) * 256;
  int i = blockIdx.x * blockDim.x + threadIdx.x;
  if (i >= total) return;
  int v    = i & 7;
  int lane = (i >> 3) & 31;
  int ks   = (i >> 8) % (K >> 5);
  int tn   = (i >> 8) / (K >> 5);
  int n = tn * 16 + (lane & 15);
  int k = (ks << 5) + (v << 1) + ((lane >> 4) << 3) + ((v >> 2) << 3);
  unsigned lo = f2bf_bits(W[(size_t)k * Nout + n]);
  unsigned hi = f2bf_bits(W[(size_t)(k + 1) * Nout + n]);
  P[i] = lo | (hi << 16);
}

// ------------------------------- SpMM scatter -------------------------------
__global__ void k_spmm_scatter(const int* __restrict__ erow, const int* __restrict__ ecol,
                               const float* __restrict__ ew, const float* __restrict__ H,
                               float* __restrict__ out, int E, int Hdim) {
  const int chunks = Hdim >> 2;
  long long tid = (long long)blockIdx.x * blockDim.x + threadIdx.x;
  if (tid >= (long long)E * chunks) return;
  int e = (int)(tid / chunks);
  int c = (int)(tid % chunks) << 2;
  int r = erow[e], s = ecol[e];
  float w = ew[e];
  const float4 h = *(const float4*)(H + (size_t)s * Hdim + c);
  float* o = out + (size_t)r * Hdim + c;
  __hip_atomic_fetch_add(o + 0, w * h.x, __ATOMIC_RELAXED, __HIP_MEMORY_SCOPE_AGENT);
  __hip_atomic_fetch_add(o + 1, w * h.y, __ATOMIC_RELAXED, __HIP_MEMORY_SCOPE_AGENT);
  __hip_atomic_fetch_add(o + 2, w * h.z, __ATOMIC_RELAXED, __HIP_MEMORY_SCOPE_AGENT);
  __hip_atomic_fetch_add(o + 3, w * h.w, __ATOMIC_RELAXED, __HIP_MEMORY_SCOPE_AGENT);
}

// ---------------- WMMA bf16 GEMM: pre-packed B, 4 waves / block -------------
__global__ void __launch_bounds__(128)
k_wmma_gemm(const unsigned short* __restrict__ A, const unsigned* __restrict__ Bp,
            const float* __restrict__ bias, float* __restrict__ C,
            int M, int K, int Nout, int relu) {
  const int lane = threadIdx.x & 31;
  const int wave = threadIdx.x >> 5;
  const int Ntiles = Nout >> 4;
  const int tn = blockIdx.y * 4 + wave;
  if (tn >= Ntiles) return;                       // wave-uniform
  const int r16 = lane & 15, half = lane >> 4;
  const unsigned short* Ar = A + ((size_t)blockIdx.x * 16 + r16) * K;
  const int Ks = K >> 5;
  const uint4* Bq = (const uint4*)Bp + (((size_t)tn * Ks) * 32 + lane) * 2;
  v8f acc = {};
  for (int ks = 0; ks < Ks; ++ks) {
    FragAB a, b;
    const int kk = ks << 5;
    *(uint4*)&a.u[0] = *(const uint4*)(Ar + kk + (half << 3));        // K pairs 0..7 / 8..15
    *(uint4*)&a.u[4] = *(const uint4*)(Ar + kk + 16 + (half << 3));   // K 16..31 halves
    *(uint4*)&b.u[0] = Bq[0];
    *(uint4*)&b.u[4] = Bq[1];
    Bq += 64;                                      // 32 lanes * 2 uint4 per k-step
    if (kk + 32 < K) __builtin_prefetch(Ar + kk + 32, 0, 1);
    acc = __builtin_amdgcn_wmma_f32_16x16x32_bf16(false, a.v, false, b.v,
                                                  (short)0, acc, false, false);
  }
  const int col = tn * 16 + r16;
  const float bv = bias ? bias[col] : 0.f;
#pragma unroll
  for (int r = 0; r < 8; ++r) {
    int row = blockIdx.x * 16 + (half << 3) + r;
    if (row < M) {
      float v = acc[r] + bv;
      if (relu) v = fmaxf(v, 0.f);
      C[(size_t)row * Nout + col] = v;
    }
  }
}

// ------- adj_recon = z @ z^T : async-LDS staged tiles, 8 waves / block ------
// Block computes 16 rows x 128 cols. A tile (16x128 bf16) + B tile (128x128
// bf16) staged via global_load_async_to_lds_b128 (ASYNCcnt) then consumed with
// ds_load_b128 fragment reads.
__global__ void __launch_bounds__(256)
k_wmma_adj(const unsigned short* __restrict__ Z, float* __restrict__ out,
           int N, int Ntiles) {
  __shared__ __align__(16) unsigned short sA[16 * 128];
  __shared__ __align__(16) unsigned short sB[128 * 128];
  const int t = threadIdx.x;
  const int lane = t & 31;
  const int wave = t >> 5;
  const int tm = blockIdx.x;
  const int nbase = blockIdx.y * 128;

  // stage A tile: 256 threads x 16B (one async b128 each)
  {
    int row = t >> 4, chunk = t & 15;
    int zr = tm * 16 + row;
    if (zr >= N) zr = N - 1;
    unsigned long long ga = (unsigned long long)(Z + (size_t)zr * 128 + chunk * 8);
    unsigned lo = (unsigned)(unsigned long long)(sA + row * 128 + chunk * 8);
    asm volatile("global_load_async_to_lds_b128 %0, %1, off"
                 :: "v"(lo), "v"(ga) : "memory");
  }
  // stage B tile: 256 threads x 8 async b128
#pragma unroll
  for (int i = 0; i < 8; ++i) {
    int idx = t + (i << 8);
    int row = idx >> 4, chunk = idx & 15;
    int zr = nbase + row;
    if (zr >= N) zr = N - 1;
    unsigned long long ga = (unsigned long long)(Z + (size_t)zr * 128 + chunk * 8);
    unsigned lo = (unsigned)(unsigned long long)(sB + row * 128 + chunk * 8);
    asm volatile("global_load_async_to_lds_b128 %0, %1, off"
                 :: "v"(lo), "v"(ga) : "memory");
  }
  asm volatile("s_wait_asynccnt 0" ::: "memory");
  __syncthreads();

  const int tn = blockIdx.y * 8 + wave;
  if (tn < Ntiles) {                               // wave-uniform
    const int r16 = lane & 15, half = lane >> 4;
    const unsigned short* Ar = sA + r16 * 128;
    const unsigned short* Br = sB + (wave * 16 + r16) * 128;
    v8f acc = {};
#pragma unroll
    for (int kk = 0; kk < 128; kk += 32) {
      FragAB a, b;
      *(uint4*)&a.u[0] = *(const uint4*)(Ar + kk + (half << 3));
      *(uint4*)&a.u[4] = *(const uint4*)(Ar + kk + 16 + (half << 3));
      *(uint4*)&b.u[0] = *(const uint4*)(Br + kk + (half << 3));
      *(uint4*)&b.u[4] = *(const uint4*)(Br + kk + 16 + (half << 3));
      acc = __builtin_amdgcn_wmma_f32_16x16x32_bf16(false, a.v, false, b.v,
                                                    (short)0, acc, false, false);
    }
    const int col = tn * 16 + r16;
#pragma unroll
    for (int r = 0; r < 8; ++r) {
      int row = tm * 16 + (half << 3) + r;
      if (row < N && col < N) out[(size_t)row * N + col] = acc[r];
    }
  }
}

// ----------------------------- LayerNorm + ReLU -----------------------------
template <int C>
__global__ void k_ln_relu(const float* __restrict__ in, const float* __restrict__ g,
                          const float* __restrict__ b, unsigned short* __restrict__ obf,
                          float* __restrict__ of) {
  __shared__ float s[C];
  const int row = blockIdx.x;
  const int t = threadIdx.x;
  const float v = in[(size_t)row * C + t];
  s[t] = v; __syncthreads();
  for (int o = C >> 1; o > 0; o >>= 1) { if (t < o) s[t] += s[t + o]; __syncthreads(); }
  const float mean = s[0] * (1.f / C);
  __syncthreads();
  const float d = v - mean;
  s[t] = d * d; __syncthreads();
  for (int o = C >> 1; o > 0; o >>= 1) { if (t < o) s[t] += s[t + o]; __syncthreads(); }
  const float var = s[0] * (1.f / C);
  float y = d * rsqrtf(var + 1e-5f) * g[t] + b[t];
  y = fmaxf(y, 0.f);
  if (obf) obf[(size_t)row * C + t] = f2bf_bits(y);
  if (of)  of[(size_t)row * C + t]  = y;
}

// ------------------------------ MLP heads -----------------------------------
__global__ void __launch_bounds__(64)
k_heads(const float* __restrict__ Fm,
        const float* __restrict__ p1w, const float* __restrict__ p1b,
        const float* __restrict__ p2w, const float* __restrict__ p2b,
        const float* __restrict__ p3w, const float* __restrict__ p3b,
        const float* __restrict__ s1w, const float* __restrict__ s1b,
        const float* __restrict__ s2w, const float* __restrict__ s2b,
        const float* __restrict__ s3w, const float* __restrict__ s3b,
        const float* __restrict__ t1w, const float* __restrict__ t1b,
        const float* __restrict__ t2w, const float* __restrict__ t2b,
        float* __restrict__ pos, float* __restrict__ siz,
        float* __restrict__ tlg, float* __restrict__ tpr) {
  __shared__ float f[128], h1[64], h2[32], lg[14];
  const int row = blockIdx.x;
  const int t = threadIdx.x;
  f[t]      = Fm[(size_t)row * 128 + t];
  f[t + 64] = Fm[(size_t)row * 128 + 64 + t];
  __syncthreads();
  // pos: 128 -> 64 -> 32 -> 2 (tanh)
  { float a = p1b[t]; for (int k = 0; k < 128; ++k) a += f[k] * p1w[k * 64 + t];
    h1[t] = fmaxf(a, 0.f); }
  __syncthreads();
  if (t < 32) { float a = p2b[t]; for (int k = 0; k < 64; ++k) a += h1[k] * p2w[k * 32 + t];
    h2[t] = fmaxf(a, 0.f); }
  __syncthreads();
  if (t < 2) { float a = p3b[t]; for (int k = 0; k < 32; ++k) a += h2[k] * p3w[k * 2 + t];
    pos[(size_t)row * 2 + t] = tanhf(a); }
  __syncthreads();
  // size: 128 -> 64 -> 32 -> 2 (sigmoid*0.78+0.02)
  { float a = s1b[t]; for (int k = 0; k < 128; ++k) a += f[k] * s1w[k * 64 + t];
    h1[t] = fmaxf(a, 0.f); }
  __syncthreads();
  if (t < 32) { float a = s2b[t]; for (int k = 0; k < 64; ++k) a += h1[k] * s2w[k * 32 + t];
    h2[t] = fmaxf(a, 0.f); }
  __syncthreads();
  if (t < 2) { float a = s3b[t]; for (int k = 0; k < 32; ++k) a += h2[k] * s3w[k * 2 + t];
    siz[(size_t)row * 2 + t] = 1.f / (1.f + expf(-a)) * 0.78f + 0.02f; }
  __syncthreads();
  // type: 128 -> 64 -> 14 (softmax)
  { float a = t1b[t]; for (int k = 0; k < 128; ++k) a += f[k] * t1w[k * 64 + t];
    h1[t] = fmaxf(a, 0.f); }
  __syncthreads();
  if (t < 14) { float a = t2b[t]; for (int k = 0; k < 64; ++k) a += h1[k] * t2w[k * 14 + t];
    lg[t] = a; tlg[(size_t)row * 14 + t] = a; }
  __syncthreads();
  if (t < 14) {
    float mx = lg[0];
    for (int j = 1; j < 14; ++j) mx = fmaxf(mx, lg[j]);
    float sum = 0.f;
    for (int j = 0; j < 14; ++j) sum += expf(lg[j] - mx);
    tpr[(size_t)row * 14 + t] = expf(lg[t] - mx) / sum;
  }
}

// ============================== host launcher ===============================
extern "C" void kernel_launch(void* const* d_in, const int* in_sizes, int n_in,
                              void* d_out, int out_size, void* d_ws, size_t ws_size,
                              hipStream_t stream) {
  (void)n_in; (void)out_size; (void)ws_size;
  const float* x    = (const float*)d_in[0];
  const int*   erow = (const int*)d_in[1];
  const int*   ecol = (const int*)d_in[2];
  const float* ew   = (const float*)d_in[3];
  const float* W1   = (const float*)d_in[4];
  const float* W2   = (const float*)d_in[5];
  const float* Wm   = (const float*)d_in[6];
  const float* Wl   = (const float*)d_in[7];
  const float* fe1w = (const float*)d_in[8],  *fe1b = (const float*)d_in[9];
  const float* ln1g = (const float*)d_in[10], *ln1b = (const float*)d_in[11];
  const float* fe2w = (const float*)d_in[12], *fe2b = (const float*)d_in[13];
  const float* ln2g = (const float*)d_in[14], *ln2b = (const float*)d_in[15];
  const float* fe3w = (const float*)d_in[16], *fe3b = (const float*)d_in[17];
  const float* ln3g = (const float*)d_in[18], *ln3b = (const float*)d_in[19];
  const float* p1w = (const float*)d_in[20], *p1b = (const float*)d_in[21];
  const float* p2w = (const float*)d_in[22], *p2b = (const float*)d_in[23];
  const float* p3w = (const float*)d_in[24], *p3b = (const float*)d_in[25];
  const float* s1w = (const float*)d_in[26], *s1b = (const float*)d_in[27];
  const float* s2w = (const float*)d_in[28], *s2b = (const float*)d_in[29];
  const float* s3w = (const float*)d_in[30], *s3b = (const float*)d_in[31];
  const float* t1w = (const float*)d_in[32], *t1b = (const float*)d_in[33];
  const float* t2w = (const float*)d_in[34], *t2b = (const float*)d_in[35];

  const int F = 128, H1 = 256, HD = 256;
  const int N = in_sizes[0] / F;   // 10000 (divisible by 16)
  const int E = in_sizes[1];       // 320000

  float* out = (float*)d_out;
  const size_t nH = (size_t)N * 128;
  float* o_z    = out;
  float* o_zm   = out + nH;
  float* o_zl   = out + 2 * nH;
  float* o_adj  = out + 3 * nH;
  float* o_pos  = out + 3 * nH + (size_t)N * N;
  float* o_size = o_pos  + (size_t)N * 2;
  float* o_tlg  = o_size + (size_t)N * 2;
  float* o_tpr  = o_tlg  + (size_t)N * 14;

  // ---- workspace bump allocator ----
  char* ws = (char*)d_ws;
  size_t off = 0;
  auto alloc = [&](size_t bytes) -> void* {
    void* p = ws + off;
    off += (bytes + 255) & ~(size_t)255;
    return p;
  };
  unsigned short* xbf    = (unsigned short*)alloc((size_t)N * F * 2);
  unsigned short* h1bf   = (unsigned short*)alloc((size_t)N * H1 * 2);
  unsigned short* h2bf   = (unsigned short*)alloc((size_t)N * 128 * 2);
  unsigned short* zbf    = (unsigned short*)alloc((size_t)N * 128 * 2);
  unsigned short* f256bf = (unsigned short*)alloc((size_t)N * HD * 2);
  unsigned* pW1  = (unsigned*)alloc((size_t)F  * H1  * 2);   // K*Nout/2 dwords
  unsigned* pW2  = (unsigned*)alloc((size_t)H1 * 128 * 2);
  unsigned* pWm  = (unsigned*)alloc((size_t)128 * 128 * 2);
  unsigned* pWl  = (unsigned*)alloc((size_t)128 * 128 * 2);
  unsigned* pFe1 = (unsigned*)alloc((size_t)F  * HD  * 2);
  unsigned* pFe2 = (unsigned*)alloc((size_t)HD * HD  * 2);
  unsigned* pFe3 = (unsigned*)alloc((size_t)HD * 128 * 2);
  float* tmp256 = (float*)alloc((size_t)N * 256 * 4);
  float* acc256 = (float*)alloc((size_t)N * 256 * 4);
  float* tmp128 = (float*)alloc((size_t)N * 128 * 4);
  float* acc128 = (float*)alloc((size_t)N * 128 * 4);
  float* f_f32  = (float*)alloc((size_t)N * 128 * 4);

  auto conv = [&](const float* src, unsigned short* dst, size_t n, int relu) {
    k_to_bf16<<<dim3((unsigned)((n + 255) / 256)), dim3(256), 0, stream>>>(src, dst, n, relu);
  };
  auto packb = [&](const float* W, unsigned* P, int K, int Nout) {
    int total = (K >> 5) * (Nout >> 4) * 256;
    k_pack_b<<<dim3((unsigned)((total + 255) / 256)), dim3(256), 0, stream>>>(W, P, K, Nout);
  };
  auto gemm = [&](const unsigned short* A, const unsigned* Bp, const float* bias,
                  float* C, int M, int K, int Nout, int relu) {
    int Ntiles = Nout >> 4;
    dim3 g((unsigned)((M + 15) / 16), (unsigned)((Ntiles + 3) / 4));
    k_wmma_gemm<<<g, dim3(128), 0, stream>>>(A, Bp, bias, C, M, K, Nout, relu);
  };
  auto zero = [&](float* p, size_t n) {
    k_zero_f32<<<dim3((unsigned)((n + 255) / 256)), dim3(256), 0, stream>>>(p, n);
  };
  auto spmm = [&](const float* H, float* o, int Hdim) {
    long long tot = (long long)E * (Hdim / 4);
    k_spmm_scatter<<<dim3((unsigned)((tot + 255) / 256)), dim3(256), 0, stream>>>(
        erow, ecol, ew, H, o, E, Hdim);
  };

  // ---- convert activations / pack weights ----
  conv(x, xbf, (size_t)N * F, 0);
  packb(W1,   pW1,  F,  H1);
  packb(W2,   pW2,  H1, 128);
  packb(Wm,   pWm,  128, 128);
  packb(Wl,   pWl,  128, 128);
  packb(fe1w, pFe1, F,  HD);
  packb(fe2w, pFe2, HD, HD);
  packb(fe3w, pFe3, HD, 128);

  // ---- GCN encoder ----
  gemm(xbf, pW1, nullptr, tmp256, N, F, H1, 0);            // x @ W1
  zero(acc256, (size_t)N * H1);
  spmm(tmp256, acc256, H1);
  conv(acc256, h1bf, (size_t)N * H1, 1);                   // relu -> h1 (bf16)

  gemm(h1bf, pW2, nullptr, tmp128, N, H1, 128, 0);         // h1 @ W2
  zero(acc128, (size_t)N * 128);
  spmm(tmp128, acc128, 128);
  conv(acc128, h2bf, (size_t)N * 128, 1);                  // relu -> h2 (bf16)

  gemm(h2bf, pWm, nullptr, tmp128, N, 128, 128, 0);        // h2 @ Wm
  zero(acc128, (size_t)N * 128);
  spmm(tmp128, acc128, 128);                               // z_mean
  k_write_z<<<dim3((unsigned)((nH + 255) / 256)), dim3(256), 0, stream>>>(
      acc128, o_z, o_zm, zbf, nH);

  gemm(h2bf, pWl, nullptr, tmp128, N, 128, 128, 0);        // h2 @ Wl
  zero(acc128, (size_t)N * 128);
  spmm(tmp128, acc128, 128);
  k_clip_write<<<dim3((unsigned)((nH + 255) / 256)), dim3(256), 0, stream>>>(
      acc128, o_zl, nH);

  // ---- adjacency reconstruction: z @ z^T ----
  {
    int nt = (N + 15) / 16;
    dim3 g((unsigned)nt, (unsigned)((nt + 7) / 8));
    k_wmma_adj<<<g, dim3(256), 0, stream>>>(zbf, o_adj, N, nt);
  }

  // ---- attribute decoder ----
  gemm(xbf, pFe1, fe1b, tmp256, N, F, HD, 0);
  k_ln_relu<256><<<dim3((unsigned)N), dim3(256), 0, stream>>>(tmp256, ln1g, ln1b, f256bf, nullptr);
  gemm(f256bf, pFe2, fe2b, tmp256, N, HD, HD, 0);
  k_ln_relu<256><<<dim3((unsigned)N), dim3(256), 0, stream>>>(tmp256, ln2g, ln2b, f256bf, nullptr);
  gemm(f256bf, pFe3, fe3b, tmp128, N, HD, 128, 0);
  k_ln_relu<128><<<dim3((unsigned)N), dim3(128), 0, stream>>>(tmp128, ln3g, ln3b, nullptr, f_f32);

  // ---- pos / size / type heads ----
  k_heads<<<dim3((unsigned)N), dim3(64), 0, stream>>>(
      f_f32, p1w, p1b, p2w, p2b, p3w, p3b,
      s1w, s1b, s2w, s2b, s3w, s3b,
      t1w, t1b, t2w, t2b,
      o_pos, o_size, o_tlg, o_tpr);
}
